// DeepIRTModel_49383533969779
// MI455X (gfx1250) — matrix operand order repaired and checked by reading, not compiled, not run on hardware
//
#include <hip/hip_runtime.h>
#include <hip/hip_bf16.h>

// ---------------- CDNA5 WMMA plumbing (gfx1250, wave32) ----------------
typedef __attribute__((ext_vector_type(16))) __bf16       v16bf;
typedef __attribute__((ext_vector_type(8)))  float        v8f;
typedef __attribute__((ext_vector_type(4)))  unsigned int u32x4;
typedef __attribute__((ext_vector_type(4)))  int          i32x4;
typedef __attribute__((ext_vector_type(2)))  float        f32x2;

union Frag { v16bf v; u32x4 q[2]; };

#if defined(__has_builtin)
#if __has_builtin(__builtin_amdgcn_global_load_async_to_lds_b128)
#define USE_ASYNC 1
#else
#define USE_ASYNC 0
#endif
#if __has_builtin(__builtin_amdgcn_tanhf)
#define HAVE_TANH 1
#else
#define HAVE_TANH 0
#endif
#else
#define USE_ASYNC 0
#define HAVE_TANH 0
#endif

// float -> bf16 (RNE)
__device__ __forceinline__ unsigned short f2bf(float f) {
  union { float f; unsigned u; } x; x.f = f;
  unsigned u = x.u;
  unsigned r = u + 0x7fffu + ((u >> 16) & 1u);
  return (unsigned short)(r >> 16);
}

// fast transcendentals: native v_tanh_f32 / v_rcp_f32 (gfx1250 TRANS ops)
__device__ __forceinline__ float fast_tanh(float x) {
#if HAVE_TANH
  return __builtin_amdgcn_tanhf(x);
#else
  return tanhf(x);
#endif
}
__device__ __forceinline__ float fast_sigmoid(float x) {
  return __builtin_amdgcn_rcpf(1.f + __expf(-x));
}

// A-matrix 16x32 bf16 fragment (ISA 7.12.2): lane m = lane&15, hi = lane>>4.
// elements 0..7  -> K = ks*32 + hi*8 + 0..7 ; elements 8..15 -> +16
__device__ __forceinline__ v16bf load_fragA(const unsigned short* rowbase, int ks, int hi) {
  const unsigned short* p = rowbase + ks * 32 + hi * 8;
  Frag f;
  f.q[0] = *(const u32x4*)(p);
  f.q[1] = *(const u32x4*)(p + 16);
  return f.v;
}

// B-matrix 32x16 bf16 fragment: lane n = lane&15; elements 0..7 -> K = ks*32 + hi*8 + 0..7 ; elements 8..15 -> +16
__device__ __forceinline__ v16bf load_fragB(const unsigned short* rowbase, int ks, int hi) {
  const unsigned short* p = rowbase + ks * 32 + hi * 8;
  Frag f;
  f.q[0] = *(const u32x4*)(p);
  f.q[1] = *(const u32x4*)(p + 16);
  return f.v;
}

__device__ __forceinline__ v8f wmma_bf16(v16bf a, v16bf b, v8f c) {
  return __builtin_amdgcn_wmma_f32_16x16x32_bf16(false, a, false, b, (short)0, c, false, false);
}

// ---------------- problem constants ----------------
#define NB   256
#define NS   1024
#define MM   50
#define KD   50
#define VD   200
#define FCN  50
// padded tile sizes
#define KQP  64    // q K-pad   (2 k-steps)
#define NMP  64    // mem N-pad (4 n-tiles)
#define VKP  224   // qa K-pad  (7 k-steps)
#define VNP  208   // v N-pad   (13 n-tiles)
#define PINP 256   // pin K-pad (8 k-steps)
#define FCP  64    // fc N-pad  (4 n-tiles)
// gather chunking: 16B chunks; q row = 8 chunks, qa row = 28 chunks
#define NCHUNK (16 * 8 + 16 * 28)   // 576

// ---- workspace layout (d_ws) ----
#define MV_ELEMS   ((size_t)NB * MM * VD)            // 2,560,000 f32
#define QBF_ELEMS  ((size_t)10001 * KQP)             // bf16
#define QABF_ELEMS ((size_t)20001 * VKP)             // bf16

__global__ void mv_init_kernel(const float* __restrict__ ivm, float* __restrict__ Mv) {
  size_t i = (size_t)blockIdx.x * blockDim.x + threadIdx.x;
  if (i < MV_ELEMS) Mv[i] = ivm[i % (MM * VD)];
}

__global__ void qemb_bf16_kernel(const float* __restrict__ qe, unsigned short* __restrict__ qbf) {
  size_t i = (size_t)blockIdx.x * blockDim.x + threadIdx.x;
  if (i < QBF_ELEMS) {
    int n = (int)(i >> 6), k = (int)(i & 63);
    qbf[i] = f2bf(k < KD ? qe[(size_t)n * KD + k] : 0.f);
  }
}

__global__ void qaemb_bf16_kernel(const float* __restrict__ qae, unsigned short* __restrict__ qabf) {
  size_t i = (size_t)blockIdx.x * blockDim.x + threadIdx.x;
  if (i < QABF_ELEMS) {
    int n = (int)(i / VKP), k = (int)(i - (size_t)n * VKP);
    qabf[i] = f2bf(k < VD ? qae[(size_t)n * VD + k] : 0.f);
  }
}

__global__ void zero_tail_kernel(float* __restrict__ out) {
  size_t i = (size_t)blockIdx.x * blockDim.x + threadIdx.x;
  if (i < (size_t)3 * NB * NS) out[(size_t)NB * NS + i] = 0.f;
}

// issue the embedding gather for step t: 576 x 16B chunks, bf16 tables -> LDS (ASYNCcnt)
__device__ __forceinline__ void issue_gather(
    int t, int c0, int cstride, int b0,
    const int* __restrict__ q_data, const int* __restrict__ qa_data,
    const unsigned short* __restrict__ qbf, const unsigned short* __restrict__ qabf,
    unsigned short* s_q, unsigned short* s_qa) {
  for (int c = c0; c < NCHUNK; c += cstride) {
    const unsigned short* g;
    unsigned short* l;
    if (c < 128) {
      int b = c >> 3, ch = (c & 7) * 8;
      int qi = min(max(q_data[(size_t)(b0 + b) * NS + t], 0), 10000);
      g = qbf + (size_t)qi * KQP + ch;
      l = s_q + b * KQP + ch;
    } else {
      int j = c - 128;
      int b = j / 28, ch = (j - b * 28) * 8;
      int qi = min(max(qa_data[(size_t)(b0 + b) * NS + t], 0), 20000);
      g = qabf + (size_t)qi * VKP + ch;
      l = s_qa + b * VKP + ch;
    }
#if USE_ASYNC
    __builtin_amdgcn_global_load_async_to_lds_b128((i32x4*)g, (i32x4*)l, 0, 0);
#else
    *(u32x4*)l = *(const u32x4*)g;
#endif
  }
}

// one erase/add WMMA tile-job (j in [0,26): 0-12 erase, 13-25 add)
__device__ __forceinline__ void ea_job(
    int j, int ln, int hi,
    const unsigned short* s_qa, const unsigned short* s_ew, const unsigned short* s_aw,
    const float* s_eb, const float* s_ab, float* s_er, float* s_ad) {
  int isAdd = (j >= 13);
  int nt = isAdd ? j - 13 : j;
  const unsigned short* wt = isAdd ? s_aw : s_ew;
  const float* bias        = isAdd ? s_ab : s_eb;
  float* dst               = isAdd ? s_ad : s_er;
  v8f c = {};
  const unsigned short* arow = s_qa + (unsigned)ln * VKP;
  const unsigned short* brow = wt + (unsigned)(nt * 16 + ln) * VKP;
#pragma unroll
  for (int ks = 0; ks < 7; ++ks)
    c = wmma_bf16(load_fragA(arow, ks, hi), load_fragB(brow, ks, hi), c);
  int n = nt * 16 + ln;
  float bn = bias[n];
#pragma unroll
  for (int r = 0; r < 8; ++r) {
    float x = c[r] + bn;
    float y = isAdd ? fast_tanh(x) : fast_sigmoid(x);
    dst[(r + hi * 8) * VNP + n] = y;
  }
}

// One workgroup (256 threads = 8 wave32) owns a 16-row batch tile for all 1024 steps.
__global__ __launch_bounds__(256)
void dkvmn_kernel(const int*   __restrict__ q_data,
                  const int*   __restrict__ qa_data,
                  const unsigned short* __restrict__ qbf,
                  const unsigned short* __restrict__ qabf,
                  const float* __restrict__ key_memory,
                  const float* __restrict__ erase_w,
                  const float* __restrict__ erase_b,
                  const float* __restrict__ add_w,
                  const float* __restrict__ add_b,
                  const float* __restrict__ pred_w1,
                  const float* __restrict__ pred_b1,
                  const float* __restrict__ pred_w2,
                  const float* __restrict__ pred_b2,
                  float* __restrict__ Mv,
                  float* __restrict__ out) {
  // ---- LDS: ~275 KB of the 320 KB WGP LDS ----
  __shared__ __align__(16) unsigned short s_key[NMP * KQP];
  __shared__ __align__(16) unsigned short s_ew[VNP * VKP];
  __shared__ __align__(16) unsigned short s_aw[VNP * VKP];
  __shared__ __align__(16) unsigned short s_pw1[FCP * PINP];
  __shared__ __align__(16) unsigned short s_q[16 * KQP];
  __shared__ __align__(16) unsigned short s_qa[16 * VKP];
  __shared__ __align__(16) unsigned short s_pin[16 * PINP];
  __shared__ __align__(16) float s_w[16 * NMP];
  __shared__ __align__(16) float s_er[16 * VNP];
  __shared__ __align__(16) float s_ad[16 * VNP];
  __shared__ __align__(16) float s_h[16 * FCP];
  __shared__ float s_eb[VNP], s_ab[VNP], s_pb1[FCP], s_pw2f[FCP];

  const int tid  = threadIdx.x;
  const int wave = tid >> 5;
  const int lane = tid & 31;
  const int hi   = lane >> 4;
  const int ln   = lane & 15;
  const int b0   = blockIdx.x * 16;

  // ---- one-time weight staging into LDS (bf16, zero-padded) ----
  for (int i = tid; i < NMP * KQP; i += 256) {
    int n = i >> 6, k = i & 63;
    s_key[i] = f2bf((n < MM && k < KD) ? key_memory[n * KD + k] : 0.f);
  }
  for (int i = tid; i < VNP * VKP; i += 256) {
    int n = i / VKP, k = i - n * VKP;
    bool in = (n < VD && k < VD);
    s_ew[i] = f2bf(in ? erase_w[n * VD + k] : 0.f);
    s_aw[i] = f2bf(in ? add_w[n * VD + k]   : 0.f);
  }
  for (int i = tid; i < FCP * PINP; i += 256) {
    int n = i >> 8, k = i & 255;
    s_pw1[i] = f2bf((n < FCN && k < (VD + KD)) ? pred_w1[n * (VD + KD) + k] : 0.f);
  }
  if (tid < VNP) { s_eb[tid] = tid < VD ? erase_b[tid] : 0.f;
                   s_ab[tid] = tid < VD ? add_b[tid]   : 0.f; }
  if (tid < FCP) { s_pb1[tid]  = tid < FCN ? pred_b1[tid] : 0.f;
                   s_pw2f[tid] = tid < FCN ? pred_w2[tid] : 0.f; }
  const float pb2 = pred_b2[0];

  // prologue: kick off gather for t=0 (all threads)
  issue_gather(0, tid, 256, b0, q_data, qa_data, qbf, qabf, s_q, s_qa);

  for (int t = 0; t < NS; ++t) {
    // retire the in-flight gather (per-wave ASYNCcnt), then make it WG-visible
#if USE_ASYNC
    asm volatile("s_wait_asynccnt 0" ::: "memory");
#endif
    __syncthreads();

    // phase 2: logits WMMA (waves 0-3) || erase/add jobs 0-11 (waves 4-7)
    if (wave < 4) {
      v8f c = {};
      const unsigned short* arow = s_q + (unsigned)ln * KQP;
      const unsigned short* brow = s_key + (unsigned)(wave * 16 + ln) * KQP;
#pragma unroll
      for (int ks = 0; ks < 2; ++ks)
        c = wmma_bf16(load_fragA(arow, ks, hi), load_fragB(brow, ks, hi), c);
      int n = wave * 16 + ln;
#pragma unroll
      for (int r = 0; r < 8; ++r) s_w[(r + hi * 8) * NMP + n] = c[r];
    } else {
      for (int j = wave - 4; j < 12; j += 4)
        ea_job(j, ln, hi, s_qa, s_ew, s_aw, s_eb, s_ab, s_er, s_ad);
    }
    __syncthreads();

    // phase 3: softmax (wave 0) || erase/add jobs 12-25 (waves 1-7)
    if (wave == 0) {
      if (lane < 16) {
        float mx = -1e30f;
        for (int m = 0; m < MM; ++m) mx = fmaxf(mx, s_w[lane * NMP + m]);
        float sum = 0.f;
        for (int m = 0; m < MM; ++m) {
          float e = __expf(s_w[lane * NMP + m] - mx);
          sum += e;
          s_w[lane * NMP + m] = e;
        }
        float inv = __builtin_amdgcn_rcpf(sum);
        for (int m = 0; m < MM; ++m) s_w[lane * NMP + m] *= inv;
      }
    } else {
      for (int j = 12 + (wave - 1); j < 26; j += 7)
        ea_job(j, ln, hi, s_qa, s_ew, s_aw, s_eb, s_ab, s_er, s_ad);
    }
    __syncthreads();

    // phase 4: fused memory read + update, 2-wide along V (b64 global, packed bf16 out)
    for (int idx = tid; idx < 16 * (VD / 2); idx += 256) {
      int b = idx / (VD / 2), p = idx - b * (VD / 2);
      int v = p * 2;
      f32x2 e2 = *(const f32x2*)(s_er + b * VNP + v);
      f32x2 a2 = *(const f32x2*)(s_ad + b * VNP + v);
      const float* wr = s_w + b * NMP;
      float* mp = Mv + ((size_t)(b0 + b) * MM) * VD + v;
      f32x2 acc = {0.f, 0.f};
      for (int m = 0; m < MM; ++m) {
        f32x2 mv = *(const f32x2*)(mp + m * VD);
        float wm = wr[m];
        acc.x = fmaf(wm, mv.x, acc.x);
        acc.y = fmaf(wm, mv.y, acc.y);
        mv.x = mv.x * (1.f - wm * e2.x) + wm * a2.x;
        mv.y = mv.y * (1.f - wm * e2.y) + wm * a2.y;
        *(f32x2*)(mp + m * VD) = mv;
      }
      unsigned pk = (unsigned)f2bf(acc.x) | ((unsigned)f2bf(acc.y) << 16);
      *(unsigned*)(s_pin + b * PINP + v) = pk;
    }
    for (int i = tid; i < 16 * 56; i += 256) {   // pin cols 200..255 = [q | 0 pad]
      int b = i / 56, k = i - b * 56;
      s_pin[b * PINP + VD + k] = (k < KD) ? s_q[b * KQP + k] : (unsigned short)0;
    }
    __syncthreads();

    // phase 5: predictor fc1 WMMA (waves 0-3) || async gather for t+1 (waves 4-7)
    if (wave < 4) {
      v8f c = {};
      const unsigned short* arow = s_pin + (unsigned)ln * PINP;
      const unsigned short* brow = s_pw1 + (unsigned)(wave * 16 + ln) * PINP;
#pragma unroll
      for (int ks = 0; ks < 8; ++ks)
        c = wmma_bf16(load_fragA(arow, ks, hi), load_fragB(brow, ks, hi), c);
      int n = wave * 16 + ln;
      float bn = s_pb1[n];
#pragma unroll
      for (int r = 0; r < 8; ++r) s_h[(r + hi * 8) * FCP + n] = fast_tanh(c[r] + bn);
    } else if (t + 1 < NS) {
      issue_gather(t + 1, tid - 128, 128, b0, q_data, qa_data, qbf, qabf, s_q, s_qa);
    }
    __syncthreads();

    // phase 6: fc2 + sigmoid -> preds[b, t]
    if (tid < 16) {
      float acc = pb2;
      for (int f = 0; f < FCN; ++f) acc = fmaf(s_h[tid * FCP + f], s_pw2f[f], acc);
      out[(size_t)(b0 + tid) * NS + t] = fast_sigmoid(acc);
    }
    __syncthreads();
  }
}

extern "C" void kernel_launch(void* const* d_in, const int* in_sizes, int n_in,
                              void* d_out, int out_size, void* d_ws, size_t ws_size,
                              hipStream_t stream) {
  const int*   q_data     = (const int*)  d_in[0];
  const int*   qa_data    = (const int*)  d_in[1];
  const float* q_embed_w  = (const float*)d_in[2];
  const float* qa_embed_w = (const float*)d_in[3];
  const float* key_memory = (const float*)d_in[4];
  const float* init_vm    = (const float*)d_in[5];
  const float* erase_w    = (const float*)d_in[6];
  const float* erase_b    = (const float*)d_in[7];
  const float* add_w      = (const float*)d_in[8];
  const float* add_b      = (const float*)d_in[9];
  const float* pred_w1    = (const float*)d_in[10];
  const float* pred_b1    = (const float*)d_in[11];
  const float* pred_w2    = (const float*)d_in[12];
  const float* pred_b2    = (const float*)d_in[13];
  float* out = (float*)d_out;

  // workspace: [Mv f32 10.24MB][q_emb bf16 1.28MB][qa_emb bf16 8.96MB]
  float* Mv = (float*)d_ws;
  unsigned short* qbf  = (unsigned short*)((char*)d_ws + MV_ELEMS * sizeof(float));
  unsigned short* qabf = qbf + QBF_ELEMS;

  mv_init_kernel<<<(unsigned)((MV_ELEMS + 255) / 256), 256, 0, stream>>>(init_vm, Mv);
  qemb_bf16_kernel<<<(unsigned)((QBF_ELEMS + 255) / 256), 256, 0, stream>>>(q_embed_w, qbf);
  qaemb_bf16_kernel<<<(unsigned)((QABF_ELEMS + 255) / 256), 256, 0, stream>>>(qa_embed_w, qabf);
  zero_tail_kernel<<<(unsigned)(((size_t)3 * NB * NS + 255) / 256), 256, 0, stream>>>(out);

  dkvmn_kernel<<<NB / 16, 256, 0, stream>>>(
      q_data, qa_data, qbf, qabf, key_memory,
      erase_w, erase_b, add_w, add_b, pred_w1, pred_b1, pred_w2, pred_b2,
      Mv, out);
}
